// MaxSigmoidAttnBlock_45707041964351
// MI455X (gfx1250) — matrix operand order repaired and checked by reading, not compiled, not run on hardware
//
#include <hip/hip_runtime.h>

#define NHEADS 8
#define HC     32
#define Bsz    8
#define C1     256
#define Him    64
#define Wim    64
#define Ng     32
#define GCc    512
#define ECc    256

typedef __attribute__((ext_vector_type(16))) __bf16 v16bf;
typedef __attribute__((ext_vector_type(8)))  float  v8f;

union Frag32B { uint4 q[2]; v16bf v; };

// CDNA5 async global->LDS path, gated so compilation never regresses.
#if defined(__gfx1250__) && \
    __has_builtin(__builtin_amdgcn_global_load_async_to_lds_b32) && \
    __has_builtin(__builtin_amdgcn_s_wait_asynccnt)
#define USE_ASYNC_LDS 1
typedef __attribute__((address_space(1))) int GASI;   // global (AS1) dword ptr
typedef __attribute__((address_space(3))) int LASI;   // LDS (AS3) dword ptr
#else
#define USE_ASYNC_LDS 0
#endif

// fp32 -> bf16, RNE via native cast (hardware cvt where available)
static __device__ __forceinline__ __bf16 f2bf(float f) { return (__bf16)f; }

// pack two fp32 -> packed bf16 dword (lo in [15:0], hi in [31:16])
static __device__ __forceinline__ unsigned pack2bf(float lo, float hi) {
#if defined(__gfx1250__) && __has_builtin(__builtin_amdgcn_cvt_pk_bf16_f32)
  auto v = __builtin_amdgcn_cvt_pk_bf16_f32(lo, hi);
  return __builtin_bit_cast(unsigned, v);
#else
  unsigned short l = __builtin_bit_cast(unsigned short, (__bf16)lo);
  unsigned short h = __builtin_bit_cast(unsigned short, (__bf16)hi);
  return ((unsigned)h << 16) | (unsigned)l;
#endif
}

// Build a 16x32 bf16 A-fragment from an fp32 row (32 contiguous K values).
// ISA layout: lane&15 = M row; lanes 0-15: halves 0..7 = K 0..7, 8..15 = K 16..23;
// lanes 16-31: halves 0..7 = K 8..15, 8..15 = K 24..31.
static __device__ __forceinline__ v16bf load_a_rowmajor(const float* rowPtr, int lane) {
  const int kb = (lane >> 4) * 8;
  v16bf a;
#pragma unroll
  for (int i = 0; i < 8; ++i) {
    a[i]     = f2bf(rowPtr[kb + i]);
    a[i + 8] = f2bf(rowPtr[16 + kb + i]);
  }
  return a;
}

// ---------------------------------------------------------------------------
// Kernel 1: g[b,n,e] = guide[b,n,:] . gl_w[e,:] + gl_b[e]
// M = B*N = 256 rows, K = GC = 512, N = EC = 256 cols. 16x16 tile per wave.
// ---------------------------------------------------------------------------
__global__ __launch_bounds__(256)
void guide_proj_kernel(const float* __restrict__ guide,
                       const float* __restrict__ glw,
                       const float* __restrict__ glb,
                       float* __restrict__ gBuf) {
  const int tid  = threadIdx.x;
  const int wave = tid >> 5, lane = tid & 31;
  const int tile = blockIdx.x * 8 + wave;     // 0..255
  const int rowT = (tile >> 4) * 16;          // row tile (b*32+n)
  const int colT = (tile & 15) * 16;          // e-column tile

  const int mrow = lane & 15;
  const int eCol = colT + mrow;
  const int cb   = (lane >> 4) * 8;
  v8f acc = {};

  for (int kc = 0; kc < GCc; kc += 32) {
    v16bf a = load_a_rowmajor(guide + (rowT + mrow) * GCc + kc, lane);
    // B[k][e]: column e fixed per lane, K cb..cb+7 then 16+cb..23+cb from gl_w row
    const float4* p4 = (const float4*)(glw + eCol * GCc + kc + cb);
    v16bf bv;
#pragma unroll
    for (int q = 0; q < 4; ++q) {
      float4 f = p4[(q & 1) + (q >> 1) * 4];
      bv[q * 4 + 0] = f2bf(f.x); bv[q * 4 + 1] = f2bf(f.y);
      bv[q * 4 + 2] = f2bf(f.z); bv[q * 4 + 3] = f2bf(f.w);
    }
    acc = __builtin_amdgcn_wmma_f32_16x16x32_bf16(false, a, false, bv,
                                                  (short)0, acc, false, false);
  }

  const int hi   = lane >> 4;
  const float bs = glb[eCol];
#pragma unroll
  for (int r = 0; r < 8; ++r) {
    const int row = rowT + r + hi * 8;
    gBuf[row * ECc + eCol] = acc[r] + bs;
  }
}

// ---------------------------------------------------------------------------
// Kernel 2: aw[b,m,s] = sigmoid(max_n <emb[b,m,:,s], g[b,n,m,:]> / sqrt(hc) + bias[m])
// Per wave: scores (32 guides x 16 spatial) via two 16x16x32 bf16 WMMAs,
// max over guide rows (in-lane + shfl_xor across lane halves), sigmoid.
// ---------------------------------------------------------------------------
__global__ __launch_bounds__(256)
void attn_weight_kernel(const float* __restrict__ x,
                        const float* __restrict__ gBuf,
                        const float* __restrict__ attn_bias,
                        float* __restrict__ awBuf) {
  const int tid  = threadIdx.x;
  const int wave = tid >> 5, lane = tid & 31;
  const int bm   = blockIdx.y;
  const int b    = bm >> 3, m = bm & 7;
  const int s0   = blockIdx.x * 128 + wave * 16;

  const int nr = lane & 15;
  v16bf a0 = load_a_rowmajor(gBuf + ((b * Ng + nr)      * ECc + m * HC), lane);
  v16bf a1 = load_a_rowmajor(gBuf + ((b * Ng + 16 + nr) * ECc + m * HC), lane);

  // B[c=32][s=16]: column s per lane, channels strided by H*W in x
  const int cb = (lane >> 4) * 8;
  const int s  = s0 + (lane & 15);
  const float* xp = x + ((size_t)(b * C1 + m * HC + cb)) * (Him * Wim) + s;
  v16bf bv;
#pragma unroll
  for (int h = 0; h < 16; ++h) bv[h] = f2bf(xp[((h & 7) + (h >> 3) * 16) * Him * Wim]);

  v8f z = {};
  v8f d0 = __builtin_amdgcn_wmma_f32_16x16x32_bf16(false, a0, false, bv, (short)0, z, false, false);
  v8f d1 = __builtin_amdgcn_wmma_f32_16x16x32_bf16(false, a1, false, bv, (short)0, z, false, false);

  float mx = fmaxf(d0[0], d1[0]);
#pragma unroll
  for (int r = 1; r < 8; ++r) mx = fmaxf(mx, fmaxf(d0[r], d1[r]));
  mx = fmaxf(mx, __shfl_xor(mx, 16, 32));   // combine the two lane halves

  const float zz = mx * 0.17677669529663689f + attn_bias[m]; // 1/sqrt(32)
  const float sg = 1.f / (1.f + __expf(-zz));
  if (lane < 16)
    awBuf[(b * NHEADS + m) * (Him * Wim) + s] = sg;
}

// ---------------------------------------------------------------------------
// Kernel 3: implicit-GEMM 3x3 conv + BN + aw gate, fused.
// WG tile: 64 co x 1 image row (64 x). 8 waves; each wave 16co x 32x (2 accs).
// K-loop: 8 ci-chunks of 32; per chunk stage weights (64x32x9, packed bf16
// dword stores) and the 3-row input slab. Input slab staging uses the CDNA5
// async global->LDS engine (ASYNCcnt) into an fp32 buffer, halo lanes zeroed
// with plain DS stores, then an LDS->LDS packed-bf16 conversion pass.
// ---------------------------------------------------------------------------
__global__ __launch_bounds__(256)
void conv_bn_gate_kernel(const float* __restrict__ x,
                         const float* __restrict__ pw,
                         const float* __restrict__ gamma,
                         const float* __restrict__ beta,
                         const float* __restrict__ mean,
                         const float* __restrict__ var,
                         const float* __restrict__ awBuf,
                         float* __restrict__ out) {
  __shared__ __attribute__((aligned(16))) unsigned short wTile[64 * 9 * 32];  // [co][tap][ci] 36.9 KB
  __shared__ __attribute__((aligned(16))) unsigned short inTile[3 * 72 * 32]; // [dy][j][ci]  13.8 KB
#if USE_ASYNC_LDS
  __shared__ __attribute__((aligned(16))) float inF32[3 * 72 * 32];           // fp32 staging 27.6 KB
#endif
  __shared__ float sInv[64], sAdd[64];

  const int tid  = threadIdx.x;
  const int wave = tid >> 5;
  const int lane = tid & 31;

  const int rowId  = blockIdx.x;        // b*H + y
  const int b      = rowId >> 6;
  const int y      = rowId & 63;
  const int coBase = blockIdx.y * 64;

  if (tid < 64) {
    const int co   = coBase + tid;
    const float iv = gamma[co] * rsqrtf(var[co] + 1e-3f);
    sInv[tid] = iv;
    sAdd[tid] = beta[co] - mean[co] * iv;
  }

  const int coW = (wave & 3) * 16;      // wave's co offset inside tile
  const int xW  = (wave >> 2) * 32;     // wave's x offset (2 frags of 16)

  v8f acc0 = {}; v8f acc1 = {};

  const int kb   = (lane >> 4) * 8;
  const int cbl  = (lane >> 4) * 16;
  const int mrow = lane & 15;

  for (int cc = 0; cc < 8; ++cc) {
    const int ciBase = cc * 32;
    __syncthreads();   // protect LDS vs previous iteration's readers

    // ---- stage weights: 64co x (16 ci-pairs) x 9tap, packed bf16x2 stores.
    // 9 consecutive threads read 18 contiguous floats (one ci-pair's taps).
    for (int e = tid; e < 64 * 16 * 9; e += 256) {
      const int tap = e % 9;
      const int t2  = e / 9;
      const int cip = t2 & 15;
      const int co  = t2 >> 4;
      const int ci  = cip * 2;
      const float w0 = pw[((coBase + co) * C1 + (ciBase + ci))     * 9 + tap];
      const float w1 = pw[((coBase + co) * C1 + (ciBase + ci + 1)) * 9 + tap];
      ((unsigned*)wTile)[(co * 9 + tap) * 16 + cip] = pack2bf(w0, w1);
    }

#if USE_ASYNC_LDS
    // ---- stage input slab fp32 via async DMA (no VGPR return path);
    // halo (gx/gy OOB) lanes write zeros with regular DS stores.
    for (int e = tid; e < 32 * 3 * 72; e += 256) {
      const int ci = e / (3 * 72);
      const int r  = e - ci * (3 * 72);
      const int dy = r / 72;
      const int j  = r - dy * 72;
      const int gx = j - 1;
      const int gy = y - 1 + dy;
      float* lp = &inF32[(dy * 72 + j) * 32 + ci];
      if (gx >= 0 && gx < Wim && gy >= 0 && gy < Him) {
        const float* gp = &x[((b * C1 + (ciBase + ci)) * Him + gy) * Wim + gx];
        __builtin_amdgcn_global_load_async_to_lds_b32(
            (GASI*)(uintptr_t)gp, (LASI*)(unsigned)(uintptr_t)lp, 0, 0);
      } else {
        *lp = 0.f;
      }
    }
    __builtin_amdgcn_s_wait_asynccnt(0);
    __syncthreads();   // all waves' async copies + zero stores visible
    // ---- LDS->LDS conversion pass: float2 -> packed bf16 dword
    for (int p = tid; p < 3 * 72 * 16; p += 256) {
      const int jd = p >> 4;            // dy*72 + j
      const int cp = p & 15;            // ci pair
      const float2 f = *(const float2*)&inF32[jd * 32 + cp * 2];
      ((unsigned*)inTile)[jd * 16 + cp] = pack2bf(f.x, f.y);
    }
#else
    // ---- fallback: direct fp32 load -> bf16 store
    for (int e = tid; e < 32 * 3 * 72; e += 256) {
      const int ci = e / (3 * 72);
      const int r  = e - ci * (3 * 72);
      const int dy = r / 72;
      const int j  = r - dy * 72;
      const int gx = j - 1;
      const int gy = y - 1 + dy;
      float v = 0.f;
      if (gx >= 0 && gx < Wim && gy >= 0 && gy < Him)
        v = x[((b * C1 + (ciBase + ci)) * Him + gy) * Wim + gx];
      inTile[(dy * 72 + j) * 32 + ci] = __builtin_bit_cast(unsigned short, f2bf(v));
    }
#endif
    if (cc + 1 < 8)   // warm L2 with next chunk's weights
      __builtin_prefetch(&pw[(coBase * C1 + (ciBase + 32)) * 9], 0, 1);
    __syncthreads();

#pragma unroll
    for (int tap = 0; tap < 9; ++tap) {
      const int dy = tap / 3, dx = tap % 3;
      // A: weights [16co x 32ci]
      const unsigned short* ap = &wTile[((coW + mrow) * 9 + tap) * 32];
      Frag32B A;
      A.q[0] = *(const uint4*)(ap + kb);
      A.q[1] = *(const uint4*)(ap + 16 + kb);
      // B frag 0: x = xW + n, LDS j = x + dx
      const unsigned short* bp0 = &inTile[(dy * 72 + (xW + mrow + dx)) * 32 + kb];
      Frag32B B0;
      B0.q[0] = ((const uint4*)bp0)[0];
      B0.q[1] = ((const uint4*)bp0)[2];
      acc0 = __builtin_amdgcn_wmma_f32_16x16x32_bf16(false, A.v, false, B0.v,
                                                     (short)0, acc0, false, false);
      // B frag 1: x = xW + 16 + n
      const unsigned short* bp1 = bp0 + 16 * 32;
      Frag32B B1;
      B1.q[0] = ((const uint4*)bp1)[0];
      B1.q[1] = ((const uint4*)bp1)[2];
      acc1 = __builtin_amdgcn_wmma_f32_16x16x32_bf16(false, A.v, false, B1.v,
                                                     (short)0, acc1, false, false);
    }
  }

  // epilogue: BN + per-head sigmoid gate, f32 stores
  const int n    = lane & 15;
  const int hi   = lane >> 4;
  const int head = (coBase + coW) >> 5;        // constant per wave fragment
  const float aw0 = awBuf[((b * NHEADS + head) * Him + y) * Wim + (xW + n)];
  const float aw1 = awBuf[((b * NHEADS + head) * Him + y) * Wim + (xW + 16 + n)];
#pragma unroll
  for (int r = 0; r < 8; ++r) {
    const int col  = coW + r + hi * 8;         // local co (D layout: M = r + 8*hi)
    const int co   = coBase + col;
    const float iv = sInv[col], ad = sAdd[col];
    float* op = &out[((b * C1 + co) * Him + y) * Wim];
    op[xW + n]      = (acc0[r] * iv + ad) * aw0;
    op[xW + 16 + n] = (acc1[r] * iv + ad) * aw1;
  }
}

// ---------------------------------------------------------------------------
extern "C" void kernel_launch(void* const* d_in, const int* in_sizes, int n_in,
                              void* d_out, int out_size, void* d_ws, size_t ws_size,
                              hipStream_t stream) {
  const float* x     = (const float*)d_in[0];
  const float* guide = (const float*)d_in[1];
  const float* glw   = (const float*)d_in[2];
  const float* glb   = (const float*)d_in[3];
  const float* abias = (const float*)d_in[4];
  const float* pw    = (const float*)d_in[5];
  const float* gamma = (const float*)d_in[6];
  const float* beta  = (const float*)d_in[7];
  const float* mean  = (const float*)d_in[8];
  const float* var   = (const float*)d_in[9];
  float* out = (float*)d_out;

  float* gBuf  = (float*)d_ws;                       // (8,32,256) = 256 KB
  float* awBuf = gBuf + Bsz * Ng * ECc;              // (8,8,64,64) = 1 MB

  guide_proj_kernel <<<32,            256, 0, stream>>>(guide, glw, glb, gBuf);
  attn_weight_kernel<<<dim3(32, 64),  256, 0, stream>>>(x, gBuf, abias, awBuf);
  conv_bn_gate_kernel<<<dim3(512, 4), 256, 0, stream>>>(x, pw, gamma, beta,
                                                        mean, var, awBuf, out);
}
